// FuzzySystemLayer_32624571580860
// MI455X (gfx1250) — matrix operand order repaired and checked by reading, not compiled, not run on hardware
//
#include <hip/hip_runtime.h>

// ---------------------------------------------------------------------------
// FuzzySystemLayer: out = normalize_rows(exp(-||x-c||^2 / (2 w^2))) @ W
// B=262144 rows, C=256 clusters, D=64 dims, O=8 outputs.
// Fused single kernel; cross-term GEMM via v_wmma_f32_16x16x32_bf16 with
// hi/lo bf16 split (A_hi*B_hi + A_hi*B_lo + A_lo*B_hi) for ~f32 accuracy.
// Round 3: software-pipelined tile loop (WMMAs of tile t+1 overlap the
// exp/normalize/accumulate epilogue of tile t, filling WMMA->VALU hazard
// slots and ds_load waits). DPP8 butterfly reduction, pk-f32 epilogue.
// ---------------------------------------------------------------------------

typedef __attribute__((ext_vector_type(16))) __bf16 bf16x16;
typedef __attribute__((ext_vector_type(8)))  float  f32x8;

#define NB 262144
#define NC 256
#define ND 64
#define NO 8
#define NT (NC / 16)       // 16 cluster tiles
#define WAVES 8
#define TPB (WAVES * 32)

// LDS layout (bytes). B-fragments use a 48B lane stride (16B-aligned, avoids
// the 4-way bank conflict a 32B stride would give on b128 reads).
#define FRAG_STRIDE 48
#define OFF_BHI 0
#define OFF_BLO (NT * 2 * 32 * FRAG_STRIDE)            // 49152
#define OFF_W   (OFF_BLO + NT * 2 * 32 * FRAG_STRIDE)  // 98304
#define OFF_CS  (OFF_W + NC * NO * 4)                  // 106496
#define SMEM_BYTES (OFF_CS + NC * 2 * 4)               // 108544

// DPP8 selector: lane i reads lane (i ^ m) within its group of 8.
#define DPP8_SEL(a0,a1,a2,a3,a4,a5,a6,a7) \
  ((a0)|((a1)<<3)|((a2)<<6)|((a3)<<9)|((a4)<<12)|((a5)<<15)|((a6)<<18)|((a7)<<21))
#define SEL_XOR1 DPP8_SEL(1,0,3,2,5,4,7,6)
#define SEL_XOR2 DPP8_SEL(2,3,0,1,6,7,4,5)
#define SEL_XOR4 DPP8_SEL(4,5,6,7,0,1,2,3)

template <int SEL>
__device__ __forceinline__ float dpp8_xor_add(float v) {
  int i = __builtin_bit_cast(int, v);
  int j = __builtin_amdgcn_mov_dpp8(i, SEL);
  return v + __builtin_bit_cast(float, j);
}

// Full reduction over the 16 lanes of each half: DPP8 for 1/2/4, DS for 8.
__device__ __forceinline__ float reduce16(float v) {
  v = dpp8_xor_add<SEL_XOR1>(v);
  v = dpp8_xor_add<SEL_XOR2>(v);
  v = dpp8_xor_add<SEL_XOR4>(v);
  v += __shfl_xor(v, 8, 32);
  return v;
}

__global__ __launch_bounds__(TPB)
void fuzzy_wmma_kernel(const float* __restrict__ inp,
                       const float* __restrict__ ctr,
                       const float* __restrict__ wid,
                       const float* __restrict__ rw,
                       float* __restrict__ out) {
  extern __shared__ char smem[];
  __bf16* sBhi = (__bf16*)(smem + OFF_BHI);
  __bf16* sBlo = (__bf16*)(smem + OFF_BLO);
  float*  sW   = (float*)(smem + OFF_W);    // [256][8]
  float*  sCS  = (float*)(smem + OFF_CS);   // [256] x {csq, log2e/(2w^2)}

  const int tid = threadIdx.x;

  // ---------------- Block setup (256 threads) ----------------
  {
    // Per-cluster ||c||^2 and folded exp scale; copy rule weights.
    const int n = tid;  // TPB == NC
    const float4* crow = (const float4*)(ctr + n * ND);
    float acc = 0.f;
#pragma unroll
    for (int q = 0; q < ND / 4; ++q) {
      float4 v = crow[q];
      acc += v.x * v.x + v.y * v.y + v.z * v.z + v.w * v.w;
    }
    float w = wid[n];
    sCS[2 * n + 0] = acc;
    sCS[2 * n + 1] = 0.7213475204444817f / (w * w);  // log2(e)/2 / w^2
    ((float4*)sW)[2 * n + 0] = ((const float4*)rw)[2 * n + 0];
    ((float4*)sW)[2 * n + 1] = ((const float4*)rw)[2 * n + 1];
  }

  // Pre-permute centers^T into WMMA B-fragment order (hi/lo bf16).
  // B[k][n] = centers[n][k]. Lane l holds column n = base + (l&15);
  // k = chunk*32 + (j>=8 ? 16 : 0) + (l>=16 ? 8 : 0) + (j&7)  (ISA table-K B layout).
  for (int idx = tid; idx < NC * ND; idx += TPB) {
    int j    = idx & 15;
    int lane = (idx >> 4) & 31;
    int ch   = (idx >> 9) & 1;
    int t    = idx >> 10;
    int n = t * 16 + (lane & 15);
    int k = ch * 32 + ((j >> 3) << 4) + ((lane >> 4) << 3) + (j & 7);
    float v = ctr[n * ND + k];
    __bf16 hi = (__bf16)v;
    __bf16 lo = (__bf16)(v - (float)hi);
    int e = ((t * 2 + ch) * 32 + lane) * (FRAG_STRIDE / 2) + j;
    sBhi[e] = hi;
    sBlo[e] = lo;
  }
  __syncthreads();

  // ---------------- Per-wave main loop: 16 rows per wave ----------------
  const int wave   = tid >> 5;
  const int lane   = tid & 31;
  const int laneHi = lane >> 4;
  const int l15    = lane & 15;
  const int rowBase = (blockIdx.x * WAVES + wave) * 16;
  const int m = rowBase + l15;

  // Load A (16x64 f32) in the ISA 16-bit A-matrix k-pattern:
  // lane<16: k in {0..7,16..23}+32c ; lane>=16: k in {8..15,24..31}+32c.
  const float* rowp = inp + (size_t)m * ND + laneHi * 8;
  bf16x16 ah[2], al[2];
  float xpart = 0.f;
#pragma unroll
  for (int c = 0; c < 2; ++c) {
    float4 r0 = *(const float4*)(rowp + c * 32 + 0);
    float4 r1 = *(const float4*)(rowp + c * 32 + 4);
    float4 r2 = *(const float4*)(rowp + c * 32 + 16);
    float4 r3 = *(const float4*)(rowp + c * 32 + 20);
    float v[16] = {r0.x, r0.y, r0.z, r0.w, r1.x, r1.y, r1.z, r1.w,
                   r2.x, r2.y, r2.z, r2.w, r3.x, r3.y, r3.z, r3.w};
#pragma unroll
    for (int j = 0; j < 16; ++j) {
      xpart += v[j] * v[j];
      __bf16 h = (__bf16)v[j];
      ah[c][j] = h;
      al[c][j] = (__bf16)(v[j] - (float)h);
    }
  }
  // ||x||^2 for row (l&15): combine the two half-rows, then broadcast the
  // 8 rows this lane needs in the C/D accumulator layout.
  float xrow = xpart + __shfl_xor(xpart, 16, 32);
  float xq[8];
#pragma unroll
  for (int r = 0; r < 8; ++r)
    xq[r] = __shfl(xrow, laneHi * 8 + r, 32);

  float  rs[8];
  float2 ac[8][4];
#pragma unroll
  for (int r = 0; r < 8; ++r) {
    rs[r] = 0.f;
#pragma unroll
    for (int q = 0; q < 4; ++q) ac[r][q] = make_float2(0.f, 0.f);
  }

  // --- pipeline helpers -----------------------------------------------
  auto ldfrag = [&](int t, bf16x16 f[4]) {
    const int fo0 = ((t * 2 + 0) * 32 + lane) * FRAG_STRIDE;
    const int fo1 = ((t * 2 + 1) * 32 + lane) * FRAG_STRIDE;
    __builtin_memcpy(&f[0], smem + OFF_BHI + fo0, 32);  // bh0
    __builtin_memcpy(&f[1], smem + OFF_BHI + fo1, 32);  // bh1
    __builtin_memcpy(&f[2], smem + OFF_BLO + fo0, 32);  // bl0
    __builtin_memcpy(&f[3], smem + OFF_BLO + fo1, 32);  // bl1
  };
  // Two independent accumulation chains (K-chunk 0 and 1): dependent WMMAs
  // are separated by an independent WMMA so hazard slots get covered.
  auto mm = [&](const bf16x16 f[4], f32x8& c0, f32x8& c1) {
    f32x8 z = {0.f, 0.f, 0.f, 0.f, 0.f, 0.f, 0.f, 0.f};
    c0 = __builtin_amdgcn_wmma_f32_16x16x32_bf16(false, ah[0], false, f[0],
                                                 (short)0, z, false, false);
    c1 = __builtin_amdgcn_wmma_f32_16x16x32_bf16(false, ah[1], false, f[1],
                                                 (short)0, z, false, false);
    c0 = __builtin_amdgcn_wmma_f32_16x16x32_bf16(false, ah[0], false, f[2],
                                                 (short)0, c0, false, false);
    c1 = __builtin_amdgcn_wmma_f32_16x16x32_bf16(false, ah[1], false, f[3],
                                                 (short)0, c1, false, false);
    c0 = __builtin_amdgcn_wmma_f32_16x16x32_bf16(false, al[0], false, f[0],
                                                 (short)0, c0, false, false);
    c1 = __builtin_amdgcn_wmma_f32_16x16x32_bf16(false, al[1], false, f[1],
                                                 (short)0, c1, false, false);
  };
  auto epi = [&](int t, const f32x8& c0, const f32x8& c1) {
    const int n = t * 16 + l15;
    float2 cs = ((const float2*)sCS)[n];
    float4 w0 = ((const float4*)sW)[n * 2 + 0];
    float4 w1 = ((const float4*)sW)[n * 2 + 1];
#pragma unroll
    for (int r = 0; r < 8; ++r) {
      float cross = c0[r] + c1[r];
      // -d^2 * scale = (2*cross - ||x||^2 - ||c||^2) * log2e/(2w^2)
      float arg = (2.0f * cross - xq[r] - cs.x) * cs.y;
      float e = __builtin_amdgcn_exp2f(arg);
      rs[r] += e;
      ac[r][0].x = fmaf(e, w0.x, ac[r][0].x);
      ac[r][0].y = fmaf(e, w0.y, ac[r][0].y);
      ac[r][1].x = fmaf(e, w0.z, ac[r][1].x);
      ac[r][1].y = fmaf(e, w0.w, ac[r][1].y);
      ac[r][2].x = fmaf(e, w1.x, ac[r][2].x);
      ac[r][2].y = fmaf(e, w1.y, ac[r][2].y);
      ac[r][3].x = fmaf(e, w1.z, ac[r][3].x);
      ac[r][3].y = fmaf(e, w1.w, ac[r][3].y);
    }
  };

  // --- software-pipelined tile loop (2x unrolled ping-pong, no copies) ---
  bf16x16 fA[4], fB[4];
  f32x8 a0, a1, b0, b1;
  ldfrag(0, fA);
  mm(fA, a0, a1);
  for (int t = 0; t < NT - 2; t += 2) {
    ldfrag(t + 1, fB);
    mm(fB, b0, b1);       // WMMAs for t+1 in flight...
    epi(t, a0, a1);       // ...while epilogue of t fills hazard slots
    ldfrag(t + 2, fA);
    mm(fA, a0, a1);
    epi(t + 1, b0, b1);
  }
  ldfrag(NT - 1, fB);
  mm(fB, b0, b1);
  epi(NT - 2, a0, a1);
  epi(NT - 1, b0, b1);

  // Reduce over the 16 lanes of each half (N dimension): DPP8 + one DS step.
#pragma unroll
  for (int r = 0; r < 8; ++r) {
    rs[r] = reduce16(rs[r]);
#pragma unroll
    for (int q = 0; q < 4; ++q) {
      ac[r][q].x = reduce16(ac[r][q].x);
      ac[r][q].y = reduce16(ac[r][q].y);
    }
  }

  // Lane (l&15)==r stores row rowBase + 8*laneHi + r (normalized).
#pragma unroll
  for (int r = 0; r < 8; ++r) {
    if (l15 == r) {
      float inv = __builtin_amdgcn_rcpf(rs[r]);
      int row = rowBase + laneHi * 8 + r;
      float4 o0 = make_float4(ac[r][0].x * inv, ac[r][0].y * inv,
                              ac[r][1].x * inv, ac[r][1].y * inv);
      float4 o1 = make_float4(ac[r][2].x * inv, ac[r][2].y * inv,
                              ac[r][3].x * inv, ac[r][3].y * inv);
      *(float4*)(out + (size_t)row * NO + 0) = o0;
      *(float4*)(out + (size_t)row * NO + 4) = o1;
    }
  }
}

extern "C" void kernel_launch(void* const* d_in, const int* in_sizes, int n_in,
                              void* d_out, int out_size, void* d_ws, size_t ws_size,
                              hipStream_t stream) {
  (void)in_sizes; (void)n_in; (void)out_size; (void)d_ws; (void)ws_size;
  const float* inp = (const float*)d_in[0];   // [B, 64]
  const float* ctr = (const float*)d_in[1];   // [256, 64]
  const float* wid = (const float*)d_in[2];   // [256]
  const float* rw  = (const float*)d_in[3];   // [256, 8]
  float* out = (float*)d_out;                 // [B, 8]

  // >64KB dynamic LDS needs the opt-in attribute (320KB/WGP on gfx1250).
  static_assert(SMEM_BYTES <= 160 * 1024, "LDS budget");
  hipFuncSetAttribute((const void*)fuzzy_wmma_kernel,
                      hipFuncAttributeMaxDynamicSharedMemorySize, SMEM_BYTES);

  dim3 grid(NB / (16 * WAVES));  // 2048 blocks x 256 threads (8 waves)
  hipLaunchKernelGGL(fuzzy_wmma_kernel, grid, dim3(TPB), SMEM_BYTES, stream,
                     inp, ctr, wid, rw, out);
}
